// SSMNetwork_26688926777537
// MI455X (gfx1250) — matrix operand order repaired and checked by reading, not compiled, not run on hardware
//
#include <hip/hip_runtime.h>

// ---------------------------------------------------------------------------
// SSM: h_t = A h_{t-1} + B x_t ; y_t = C h_t + D x_t
// Chunked parallel scan, bf16 WMMA (f32 accumulate), gfx1250 wave32.
// ---------------------------------------------------------------------------

#define BATCH 128
#define SEQ   2048
#define NDIM  64
#define LCH   32              // chunk length (A^32 via 5 squarings)
#define NCH   (SEQ / LCH)     // 64 chunks
#define NBT   (BATCH / 16)    // 8 batch tiles of 16
#define HS    80              // padded bf16 row stride in LDS h-tile

typedef __attribute__((ext_vector_type(16))) __bf16 v16bf;
typedef __attribute__((ext_vector_type(8)))  __bf16 v8bf;
typedef __attribute__((ext_vector_type(8)))  float  v8f;

__device__ __forceinline__ v8f wmma_bf16(v16bf a, v16bf b, v8f c) {
  // D(16x16,f32) = A(16x32,bf16) x B(32x16,bf16) + C
  return __builtin_amdgcn_wmma_f32_16x16x32_bf16(false, a, false, b,
                                                 (short)0, c, false, false);
}

__device__ __forceinline__ v8f zero8() {
  v8f z;
#pragma unroll
  for (int j = 0; j < 8; ++j) z[j] = 0.0f;
  return z;
}

// Load one 16x32 A-operand tile (row tile r, k tile kt) of a row-major 64x64
// f32 matrix into WMMA A-layout (lane m=l%16 holds row 16r+m; contiguous k:
// elem e -> K=32kt+16h+e, matching the B operands).
__device__ __forceinline__ v16bf load_aop(const float* __restrict__ M,
                                          int r, int kt, int lane) {
  const int m = lane & 15, h = lane >> 4;
  const float* p = M + (size_t)(r * 16 + m) * 64 + kt * 32 + h * 16;
  v16bf a;
#pragma unroll
  for (int e = 0; e < 8; ++e) a[e]     = (__bf16)p[e];
#pragma unroll
  for (int e = 0; e < 8; ++e) a[8 + e] = (__bf16)p[8 + e];
  return a;
}

// Publish state H (4 f32 C-tiles, 64 state rows x 16 batch cols) into LDS as
// bf16, layout [batch n][state k] with padded stride HS.
__device__ __forceinline__ void publish_h(__bf16* hs, const v8f* H, int lane) {
  const int n = lane & 15, h = lane >> 4;
#pragma unroll
  for (int r = 0; r < 4; ++r) {
    v8bf p;
#pragma unroll
    for (int j = 0; j < 8; ++j) p[j] = (__bf16)H[r][j];
    *(v8bf*)(hs + (size_t)n * HS + r * 16 + h * 8) = p;  // rows 16r+8h+0..7
  }
}

// Read back as two WMMA B-operand k-tiles: lane (n=l%16, h=l/16), elem e of
// k-tile kt holds H[k=32kt+16h+e][n]  -> contiguous 32B run in LDS.
__device__ __forceinline__ void load_hop(const __bf16* hs, int lane, v16bf* Hop) {
  const int n = lane & 15, h = lane >> 4;
#pragma unroll
  for (int kt = 0; kt < 2; ++kt) {
    const v8bf* p = (const v8bf*)(hs + (size_t)n * HS + kt * 32 + h * 16);
    v8bf lo = p[0], hi = p[1];
    Hop[kt] = __builtin_shufflevector(lo, hi, 0, 1, 2, 3, 4, 5, 6, 7,
                                      8, 9, 10, 11, 12, 13, 14, 15);
  }
}

// Fetch x_t for this batch tile: lane (n,h) reads 2x16 contiguous f32
// (d = 32*kt + 16h + 0..15) -> raw f32 regs (prefetchable), cvt later.
__device__ __forceinline__ void xfetch(const float* __restrict__ x, int b0,
                                       int t, int lane, float4* xr) {
  const int n = lane & 15, h = lane >> 4;
  const float* p = x + ((size_t)(b0 + n) * SEQ + t) * 64 + h * 16;
#pragma unroll
  for (int q = 0; q < 4; ++q) xr[q] = ((const float4*)p)[q];
  const float* p1 = p + 32;
#pragma unroll
  for (int q = 0; q < 4; ++q) xr[4 + q] = ((const float4*)p1)[q];
}

__device__ __forceinline__ v16bf cvt16(const float4* q) {
  v16bf v;
#pragma unroll
  for (int i = 0; i < 4; ++i) {
    v[4 * i + 0] = (__bf16)q[i].x;
    v[4 * i + 1] = (__bf16)q[i].y;
    v[4 * i + 2] = (__bf16)q[i].z;
    v[4 * i + 3] = (__bf16)q[i].w;
  }
  return v;
}

// ---------------------------------------------------------------------------
// Kernel 0: A32 = A^LCH by 5 repeated squarings (tiny, plain VALU).
// ---------------------------------------------------------------------------
__global__ void __launch_bounds__(256) ssm_powA(const float* __restrict__ A,
                                                float* __restrict__ A32) {
  __shared__ float Ma[64 * 64];
  __shared__ float Mb[64 * 64];
  const int tid = threadIdx.x;
  for (int i = tid; i < 64 * 64; i += 256) Ma[i] = A[i];
  __syncthreads();
  float* src = Ma;
  float* dst = Mb;
  for (int it = 0; it < 5; ++it) {  // A^2,4,8,16,32
    for (int e = tid * 16; e < tid * 16 + 16; ++e) {
      const int r = e >> 6, c = e & 63;
      float s = 0.0f;
#pragma unroll 8
      for (int j = 0; j < 64; ++j) s = fmaf(src[r * 64 + j], src[j * 64 + c], s);
      dst[e] = s;
    }
    __syncthreads();
    float* t = src; src = dst; dst = t;
  }
  for (int i = tid; i < 64 * 64; i += 256) A32[i] = src[i];
}

// ---------------------------------------------------------------------------
// Kernel 1: per (batch-tile, chunk) local scan from h=0; save final state f.
// ---------------------------------------------------------------------------
__global__ void __launch_bounds__(32) ssm_local(const float* __restrict__ x,
                                                const float* __restrict__ A,
                                                const float* __restrict__ B,
                                                float* __restrict__ f) {
  __shared__ alignas(16) __bf16 hs[16 * HS];
  const int lane = threadIdx.x;
  const int bt = blockIdx.x, c = blockIdx.y;
  const int t0 = c * LCH, b0 = bt * 16;

  v16bf Aop[4][2], Bop[4][2];
#pragma unroll
  for (int r = 0; r < 4; ++r)
#pragma unroll
    for (int kt = 0; kt < 2; ++kt) {
      Aop[r][kt] = load_aop(A, r, kt, lane);
      Bop[r][kt] = load_aop(B, r, kt, lane);
    }

  v8f H[4];
#pragma unroll
  for (int r = 0; r < 4; ++r) H[r] = zero8();
  publish_h(hs, H, lane);

  float4 xr[8];
  xfetch(x, b0, t0, lane, xr);

  for (int s = 0; s < LCH; ++s) {
    v16bf X0 = cvt16(xr), X1 = cvt16(xr + 4);
    if (s + 1 < LCH) xfetch(x, b0, t0 + s + 1, lane, xr);  // prefetch next t
    v16bf Hop[2];
    load_hop(hs, lane, Hop);
#pragma unroll
    for (int r = 0; r < 4; ++r) {
      v8f acc = zero8();
      acc = wmma_bf16(Bop[r][1], X1, acc);
      acc = wmma_bf16(Bop[r][0], X0, acc);
      acc = wmma_bf16(Aop[r][1], Hop[1], acc);
      acc = wmma_bf16(Aop[r][0], Hop[0], acc);
      H[r] = acc;
    }
    publish_h(hs, H, lane);
  }
#pragma unroll
  for (int r = 0; r < 4; ++r)
    *(v8f*)(f + ((size_t)(bt * NCH + c) * 4 + r) * 256 + lane * 8) = H[r];
}

// ---------------------------------------------------------------------------
// Kernel 2: chunk-boundary prefix scan: Hs[c] = A32*Hs[c-1] + f[c-1].
// ---------------------------------------------------------------------------
__global__ void __launch_bounds__(32) ssm_prefix(const float* __restrict__ P,
                                                 const float* __restrict__ f,
                                                 float* __restrict__ Hs) {
  __shared__ alignas(16) __bf16 hs[16 * HS];
  const int lane = threadIdx.x;
  const int bt = blockIdx.x;

  v16bf Pop[4][2];
#pragma unroll
  for (int r = 0; r < 4; ++r)
#pragma unroll
    for (int kt = 0; kt < 2; ++kt) Pop[r][kt] = load_aop(P, r, kt, lane);

  v8f H[4];
#pragma unroll
  for (int r = 0; r < 4; ++r) H[r] = zero8();
#pragma unroll
  for (int r = 0; r < 4; ++r)
    *(v8f*)(Hs + ((size_t)(bt * NCH) * 4 + r) * 256 + lane * 8) = H[r];

  for (int c = 1; c < NCH; ++c) {
    publish_h(hs, H, lane);
    v16bf Hop[2];
    load_hop(hs, lane, Hop);
#pragma unroll
    for (int r = 0; r < 4; ++r) {
      v8f acc = *(const v8f*)(f + ((size_t)(bt * NCH + (c - 1)) * 4 + r) * 256 +
                              lane * 8);
      acc = wmma_bf16(Pop[r][1], Hop[1], acc);
      acc = wmma_bf16(Pop[r][0], Hop[0], acc);
      H[r] = acc;
    }
#pragma unroll
    for (int r = 0; r < 4; ++r)
      *(v8f*)(Hs + ((size_t)(bt * NCH + c) * 4 + r) * 256 + lane * 8) = H[r];
  }
}

// ---------------------------------------------------------------------------
// Kernel 3: re-scan each chunk from its true start state and emit y.
// wave0: recurrence (A,B), publishes h_t and x_t to LDS (double buffered).
// wave1: y_t = C h_t + D x_t, stores y. One s_barrier per step.
// ---------------------------------------------------------------------------
__global__ void __launch_bounds__(64) ssm_out(const float* __restrict__ x,
                                              const float* __restrict__ A,
                                              const float* __restrict__ B,
                                              const float* __restrict__ C,
                                              const float* __restrict__ D,
                                              const float* __restrict__ Hs,
                                              float* __restrict__ y) {
  __shared__ alignas(16) __bf16 hs[2][16 * HS];
  __shared__ v16bf xsh[2][2][32];
  const int lane = threadIdx.x & 31;
  const int wid = threadIdx.x >> 5;
  const int bt = blockIdx.x, c = blockIdx.y;
  const int t0 = c * LCH, b0 = bt * 16;

  if (wid == 0) {
    v16bf Aop[4][2], Bop[4][2];
#pragma unroll
    for (int r = 0; r < 4; ++r)
#pragma unroll
      for (int kt = 0; kt < 2; ++kt) {
        Aop[r][kt] = load_aop(A, r, kt, lane);
        Bop[r][kt] = load_aop(B, r, kt, lane);
      }
    v8f H[4];
#pragma unroll
    for (int r = 0; r < 4; ++r)
      H[r] = *(const v8f*)(Hs + ((size_t)(bt * NCH + c) * 4 + r) * 256 +
                           lane * 8);
    publish_h(hs[1], H, lane);  // slot for s=0's "previous" state

    float4 xr[8];
    xfetch(x, b0, t0, lane, xr);

    for (int s = 0; s < LCH; ++s) {
      const int cur = s & 1;
      v16bf X0 = cvt16(xr), X1 = cvt16(xr + 4);
      xsh[cur][0][lane] = X0;
      xsh[cur][1][lane] = X1;
      if (s + 1 < LCH) xfetch(x, b0, t0 + s + 1, lane, xr);  // prefetch
      v16bf Hop[2];
      load_hop(hs[cur ^ 1], lane, Hop);
#pragma unroll
      for (int r = 0; r < 4; ++r) {
        v8f acc = zero8();
        acc = wmma_bf16(Bop[r][1], X1, acc);
        acc = wmma_bf16(Bop[r][0], X0, acc);
        acc = wmma_bf16(Aop[r][1], Hop[1], acc);
        acc = wmma_bf16(Aop[r][0], Hop[0], acc);
        H[r] = acc;
      }
      publish_h(hs[cur], H, lane);
      __syncthreads();
    }
  } else {
    v16bf Cop[4][2], Dop[4][2];
#pragma unroll
    for (int r = 0; r < 4; ++r)
#pragma unroll
      for (int kt = 0; kt < 2; ++kt) {
        Cop[r][kt] = load_aop(C, r, kt, lane);
        Dop[r][kt] = load_aop(D, r, kt, lane);
      }
    const int n = lane & 15, h = lane >> 4;
    for (int s = 0; s < LCH; ++s) {
      const int cur = s & 1;
      __syncthreads();
      v16bf Hop[2];
      load_hop(hs[cur], lane, Hop);
      v16bf X0 = xsh[cur][0][lane], X1 = xsh[cur][1][lane];
      float* yp = y + ((size_t)(b0 + n) * SEQ + (t0 + s)) * 64 + h * 8;
#pragma unroll
      for (int r = 0; r < 4; ++r) {
        v8f acc = zero8();
        acc = wmma_bf16(Dop[r][1], X1, acc);
        acc = wmma_bf16(Dop[r][0], X0, acc);
        acc = wmma_bf16(Cop[r][1], Hop[1], acc);
        acc = wmma_bf16(Cop[r][0], Hop[0], acc);
        *(v8f*)(yp + r * 16) = acc;  // y[b][t][16r+8h .. +7]
      }
    }
  }
}

// ---------------------------------------------------------------------------
extern "C" void kernel_launch(void* const* d_in, const int* in_sizes, int n_in,
                              void* d_out, int out_size, void* d_ws,
                              size_t ws_size, hipStream_t stream) {
  (void)in_sizes; (void)n_in; (void)out_size; (void)ws_size;
  const float* x = (const float*)d_in[0];
  const float* A = (const float*)d_in[1];
  const float* B = (const float*)d_in[2];
  const float* C = (const float*)d_in[3];
  const float* D = (const float*)d_in[4];
  float* y = (float*)d_out;

  float* ws  = (float*)d_ws;
  float* A32 = ws;                                   // 4096 f32
  float* f   = ws + 4096;                            // NBT*NCH*1024 f32
  float* Hsb = f + (size_t)NBT * NCH * 1024;         // NBT*NCH*1024 f32

  ssm_powA<<<1, 256, 0, stream>>>(A, A32);
  ssm_local<<<dim3(NBT, NCH), 32, 0, stream>>>(x, A, B, f);
  ssm_prefix<<<NBT, 32, 0, stream>>>(A32, f, Hsb);
  ssm_out<<<dim3(NBT, NCH), 64, 0, stream>>>(x, A, B, C, D, Hsb, y);
}
